// SimpleCrossLayerTranscoder_35029753266536
// MI455X (gfx1250) — matrix operand order repaired and checked by reading, not compiled, not run on hardware
//
#include <hip/hip_runtime.h>
#include <hip/hip_bf16.h>

// ---------------------------------------------------------------------------
// SimpleCrossLayerTranscoder forward for MI455X (gfx1250), bf16 WMMA path,
// double-buffered LDS pipeline (one barrier per k-step, global latency hidden
// under v_wmma_f32_16x16x32_bf16 + ds_load_b128 fragment feeds).
//   encode: pre[b,l,f]  = sum_a ((acts[b,l,a]-in_mean[l,a])/in_std[l,a]) * W_enc[l,a,f]
//           feat = relu(pre)
//   decode: rn[b,t,a]   = sum_{f<=t} sum_k feat[b,f,k] * W_dec[f,t,k,a]
//           rc = rn * out_std[t,a] + out_mean[t,a]
// ---------------------------------------------------------------------------

typedef __bf16 bf16_t;
typedef __attribute__((ext_vector_type(16))) __bf16 v16bf;
typedef __attribute__((ext_vector_type(8)))  __bf16 v8bf;
typedef __attribute__((ext_vector_type(8)))  float  v8f;
typedef __attribute__((ext_vector_type(4)))  float  v4f;

#define B_SZ 1024
#define NL   8
#define DA   768
#define DF   3072
#define KS_D (DF / 32)   // 96 k-steps per from-layer in decode

// round-to-nearest-even f32 -> bf16
__device__ __forceinline__ bf16_t f2bf(float f) {
  unsigned u = __builtin_bit_cast(unsigned, f);
  unsigned r = u + 0x7FFFu + ((u >> 16) & 1u);
  unsigned short h = (unsigned short)(r >> 16);
  return __builtin_bit_cast(bf16_t, h);
}

// Per-thread staging registers for one 128x32 A tile slice + 32x128 B tile slice
struct Regs {
  v8bf a[2];     // two (m, 8k) chunks in fragment order
  v8bf b0, b1;   // one B fragment lane: 16 bf16
};

// ---------------------------------------------------------------------------
// Encode: per layer l, [1024 x 768] @ [768 x 3072]  (standardize A, relu out)
// Block tile 128x128, BK=32, 256 threads (8 waves, 4Mx2N), wave tile 32x64.
// LDS in WMMA fragment order (ISA 7.12.2):
//   A: lane = m%16 + 16*khalf, elem i -> K = i%8 + 16*(i/8) + 8*khalf
//   B: lane = n%16 + 16*khalf, elem i -> K = i%8 + 16*(i/8) + 8*khalf
// ---------------------------------------------------------------------------
__global__ __launch_bounds__(256) void enc_kernel(
    const float* __restrict__ acts, const float* __restrict__ Wenc,
    const float* __restrict__ in_mean, const float* __restrict__ in_std,
    float* __restrict__ pre, float* __restrict__ feat)
{
  __shared__ __align__(32) bf16_t lA[2][8 * 32 * 16];   // 2 x 8 KB
  __shared__ __align__(32) bf16_t lB[2][8 * 32 * 16];   // 2 x 8 KB

  const int tid   = threadIdx.x;
  const int lane  = tid & 31;
  const int wid   = tid >> 5;
  const int waveM = wid & 3;     // 0..3 -> rows of 32
  const int waveN = wid >> 2;    // 0..1 -> cols of 64
  const int m0 = blockIdx.x * 128;
  const int n0 = blockIdx.y * 128;
  const int l  = blockIdx.z;

  const float* Bsrc = Wenc + (size_t)l * DA * DF;       // [DA, DF] row-major
  const int g      = tid & 3;                           // A-fill k-group (both tasks)
  const int nB     = n0 + (wid << 4) + (lane & 15);     // B-fill column
  const int kBhalf = (lane >> 4) << 3;                  // B-fill k sub-offset

  // -------- tile fetch into registers (global -> VGPR, cvt to bf16) --------
  auto fetchT = [&](int k0, Regs& r) {
    const float* mu = in_mean + (size_t)l * DA + k0 + g * 8;
    const float* sd = in_std  + (size_t)l * DA + k0 + g * 8;
    v4f mu0 = *(const v4f*)mu, mu1 = *(const v4f*)(mu + 4);
    v4f sd0 = *(const v4f*)sd, sd1 = *(const v4f*)(sd + 4);
#pragma unroll
    for (int j = 0; j < 2; ++j) {
      int m = (tid + j * 256) >> 2;                     // 0..127
      const float* src = acts + ((size_t)(m0 + m) * NL + l) * DA + k0 + g * 8;
      v4f x0 = *(const v4f*)src, x1 = *(const v4f*)(src + 4);
      v8bf t;
#pragma unroll
      for (int i = 0; i < 4; ++i) t[i]     = f2bf((x0[i] - mu0[i]) / sd0[i]);
#pragma unroll
      for (int i = 0; i < 4; ++i) t[4 + i] = f2bf((x1[i] - mu1[i]) / sd1[i]);
      r.a[j] = t;
    }
    const float* bp = Bsrc + (size_t)(k0 + kBhalf) * DF + nB;
    v8bf t0, t1;
#pragma unroll
    for (int i = 0; i < 8; ++i) t0[i] = f2bf(bp[(size_t)i * DF]);
#pragma unroll
    for (int i = 0; i < 8; ++i) t1[i] = f2bf(bp[(size_t)(16 + i) * DF]);
    r.b0 = t0; r.b1 = t1;
    if (k0 + 64 < DA)
      __builtin_prefetch(bp + (size_t)64 * DF, 0, 0);   // global_prefetch_b8
  };
  // -------- commit staged registers to LDS buffer (fragment order) --------
  auto storeT = [&](int buf, const Regs& r) {
#pragma unroll
    for (int j = 0; j < 2; ++j) {
      int m = (tid + j * 256) >> 2;
      int alane = (m & 15) + ((g & 1) << 4);
      *(v8bf*)&lA[buf][(((m >> 4) * 32 + alane) << 4) + ((g >> 1) << 3)] = r.a[j];
    }
    bf16_t* dst = &lB[buf][tid << 4];
    *(v8bf*)dst       = r.b0;
    *(v8bf*)(dst + 8) = r.b1;
  };

  const v8f vzero = {0.f, 0.f, 0.f, 0.f, 0.f, 0.f, 0.f, 0.f};
  v8f acc[2][4];
#pragma unroll
  for (int mt = 0; mt < 2; ++mt)
#pragma unroll
    for (int nt = 0; nt < 4; ++nt) acc[mt][nt] = vzero;

  { Regs r; fetchT(0, r); storeT(0, r); }
  int cur = 0;
  for (int k0 = 0; k0 < DA; k0 += 32) {
    __syncthreads();                                    // lds[cur] ready
    const bool more = (k0 + 32) < DA;
    Regs rnx;
    if (more) fetchT(k0 + 32, rnx);                     // in flight under WMMA

    v16bf afrag[2], bfrag[4];
#pragma unroll
    for (int mt = 0; mt < 2; ++mt)
      afrag[mt] = *(const v16bf*)&lA[cur][((waveM * 2 + mt) * 32 + lane) << 4];
#pragma unroll
    for (int nt = 0; nt < 4; ++nt)
      bfrag[nt] = *(const v16bf*)&lB[cur][((waveN * 4 + nt) * 32 + lane) << 4];
#pragma unroll
    for (int mt = 0; mt < 2; ++mt)
#pragma unroll
      for (int nt = 0; nt < 4; ++nt)
        acc[mt][nt] = __builtin_amdgcn_wmma_f32_16x16x32_bf16(
            false, afrag[mt], false, bfrag[nt], (short)0, acc[mt][nt],
            false, false);

    if (more) storeT(cur ^ 1, rnx);                     // other buffer: no hazard
    cur ^= 1;
  }

  // ---- epilogue: C/D layout VGPR r -> M = r + 8*(lane/16), N = lane%16 ----
  const int half = lane >> 4;
  const int ncol = lane & 15;
#pragma unroll
  for (int mt = 0; mt < 2; ++mt)
#pragma unroll
    for (int nt = 0; nt < 4; ++nt)
#pragma unroll
      for (int r = 0; r < 8; ++r) {
        int m = m0 + (waveM * 2 + mt) * 16 + r + half * 8;
        int n = n0 + (waveN * 4 + nt) * 16 + ncol;
        size_t idx = ((size_t)m * NL + l) * DF + n;
        float v = acc[mt][nt][r];
        pre[idx]  = v;
        feat[idx] = v > 0.f ? v : 0.f;
      }
}

// ---------------------------------------------------------------------------
// Decode: per to-layer t, sum over f<=t of [1024 x 3072] @ [3072 x 768].
// (f, k0) nest flattened into one step counter so the double-buffer pipeline
// runs across from-layer boundaries. Fused affine de-normalization.
// ---------------------------------------------------------------------------
__global__ __launch_bounds__(256) void dec_kernel(
    const float* __restrict__ feat, const float* __restrict__ Wdec,
    const float* __restrict__ out_mean, const float* __restrict__ out_std,
    float* __restrict__ rn, float* __restrict__ rc)
{
  __shared__ __align__(32) bf16_t lA[2][8 * 32 * 16];
  __shared__ __align__(32) bf16_t lB[2][8 * 32 * 16];

  const int tid   = threadIdx.x;
  const int lane  = tid & 31;
  const int wid   = tid >> 5;
  const int waveM = wid & 3;
  const int waveN = wid >> 2;
  const int m0 = blockIdx.x * 128;
  const int n0 = blockIdx.y * 128;
  const int t  = blockIdx.z;

  const int g      = tid & 3;
  const int nB     = n0 + (wid << 4) + (lane & 15);
  const int kBhalf = (lane >> 4) << 3;

  auto fetchT = [&](int s, Regs& r) {
    int f  = s / KS_D;
    int k0 = (s - f * KS_D) * 32;
#pragma unroll
    for (int j = 0; j < 2; ++j) {
      int m = (tid + j * 256) >> 2;
      const float* src = feat + ((size_t)(m0 + m) * NL + f) * DF + k0 + g * 8;
      v4f x0 = *(const v4f*)src, x1 = *(const v4f*)(src + 4);
      v8bf tv;
#pragma unroll
      for (int i = 0; i < 4; ++i) tv[i]     = f2bf(x0[i]);
#pragma unroll
      for (int i = 0; i < 4; ++i) tv[4 + i] = f2bf(x1[i]);
      r.a[j] = tv;
    }
    const float* Bsrc = Wdec + ((size_t)(f * NL + t)) * DF * DA;   // [DF, DA]
    const float* bp = Bsrc + (size_t)(k0 + kBhalf) * DA + nB;
    v8bf t0, t1;
#pragma unroll
    for (int i = 0; i < 8; ++i) t0[i] = f2bf(bp[(size_t)i * DA]);
#pragma unroll
    for (int i = 0; i < 8; ++i) t1[i] = f2bf(bp[(size_t)(16 + i) * DA]);
    r.b0 = t0; r.b1 = t1;
    __builtin_prefetch(bp + (size_t)32 * DA, 0, 0);     // global_prefetch_b8
  };
  auto storeT = [&](int buf, const Regs& r) {
#pragma unroll
    for (int j = 0; j < 2; ++j) {
      int m = (tid + j * 256) >> 2;
      int alane = (m & 15) + ((g & 1) << 4);
      *(v8bf*)&lA[buf][(((m >> 4) * 32 + alane) << 4) + ((g >> 1) << 3)] = r.a[j];
    }
    bf16_t* dst = &lB[buf][tid << 4];
    *(v8bf*)dst       = r.b0;
    *(v8bf*)(dst + 8) = r.b1;
  };

  const v8f vzero = {0.f, 0.f, 0.f, 0.f, 0.f, 0.f, 0.f, 0.f};
  v8f acc[2][4];
#pragma unroll
  for (int mt = 0; mt < 2; ++mt)
#pragma unroll
    for (int nt = 0; nt < 4; ++nt) acc[mt][nt] = vzero;

  const int steps = (t + 1) * KS_D;                     // triangular mask, structural
  { Regs r; fetchT(0, r); storeT(0, r); }
  int cur = 0;
  for (int s = 0; s < steps; ++s) {
    __syncthreads();
    const bool more = (s + 1) < steps;
    Regs rnx;
    if (more) fetchT(s + 1, rnx);

    v16bf afrag[2], bfrag[4];
#pragma unroll
    for (int mt = 0; mt < 2; ++mt)
      afrag[mt] = *(const v16bf*)&lA[cur][((waveM * 2 + mt) * 32 + lane) << 4];
#pragma unroll
    for (int nt = 0; nt < 4; ++nt)
      bfrag[nt] = *(const v16bf*)&lB[cur][((waveN * 4 + nt) * 32 + lane) << 4];
#pragma unroll
    for (int mt = 0; mt < 2; ++mt)
#pragma unroll
      for (int nt = 0; nt < 4; ++nt)
        acc[mt][nt] = __builtin_amdgcn_wmma_f32_16x16x32_bf16(
            false, afrag[mt], false, bfrag[nt], (short)0, acc[mt][nt],
            false, false);

    if (more) storeT(cur ^ 1, rnx);
    cur ^= 1;
  }

  // ---- epilogue: recons_norm and recons = rn*out_std + out_mean ----
  const int half = lane >> 4;
  const int ncol = lane & 15;
#pragma unroll
  for (int mt = 0; mt < 2; ++mt)
#pragma unroll
    for (int nt = 0; nt < 4; ++nt)
#pragma unroll
      for (int r = 0; r < 8; ++r) {
        int m = m0 + (waveM * 2 + mt) * 16 + r + half * 8;
        int n = n0 + (waveN * 4 + nt) * 16 + ncol;
        size_t idx = ((size_t)m * NL + t) * DA + n;
        float v = acc[mt][nt][r];
        rn[idx] = v;
        rc[idx] = v * out_std[(size_t)t * DA + n] + out_mean[(size_t)t * DA + n];
      }
}

// ---------------------------------------------------------------------------
extern "C" void kernel_launch(void* const* d_in, const int* in_sizes, int n_in,
                              void* d_out, int out_size, void* d_ws, size_t ws_size,
                              hipStream_t stream) {
  (void)in_sizes; (void)n_in; (void)out_size; (void)d_ws; (void)ws_size;
  const float* acts     = (const float*)d_in[0];
  const float* Wenc     = (const float*)d_in[1];
  const float* Wdec     = (const float*)d_in[2];
  // d_in[3] = mask: enforced structurally (f <= t loop), unused here.
  const float* in_mean  = (const float*)d_in[4];
  const float* in_std   = (const float*)d_in[5];
  const float* out_mean = (const float*)d_in[6];
  const float* out_std  = (const float*)d_in[7];

  float* out  = (float*)d_out;
  float* pre  = out;                                  // [B, L, DF]
  float* feat = pre  + (size_t)B_SZ * NL * DF;        // [B, L, DF]
  float* rn   = feat + (size_t)B_SZ * NL * DF;        // [B, L, DA]
  float* rc   = rn   + (size_t)B_SZ * NL * DA;        // [B, L, DA]

  dim3 ge(B_SZ / 128, DF / 128, NL);   // 8 x 24 x 8
  enc_kernel<<<ge, 256, 0, stream>>>(acts, Wenc, in_mean, in_std, pre, feat);

  dim3 gd(B_SZ / 128, DA / 128, NL);   // 8 x 6 x 8
  dec_kernel<<<gd, 256, 0, stream>>>(feat, Wdec, out_mean, out_std, rn, rc);
}
